// AgriMatcher_88897233093009
// MI455X (gfx1250) — matrix operand order repaired and checked by reading, not compiled, not run on hardware
//
#include <hip/hip_runtime.h>
#include <math.h>

#define NPTS   4096
#define CFEAT  128
#define EPSLN  1e-5f
#define REGL   1e-4f

#define LDA0 264   // stride (halves) of big LDS activation buffer (64 x 256 used)
#define LDA1 136   // stride (halves) of small LDS activation buffer (64 x 128 used)

typedef __attribute__((ext_vector_type(16))) _Float16 v16h;
typedef __attribute__((ext_vector_type(8)))  _Float16 v8h_t;
typedef __attribute__((ext_vector_type(8)))  float    v8f;

// Explicit LDS (address_space(3)) pointer types: one addrspacecast at kernel
// entry, then pure 32-bit DS address arithmetic with immediate offsets.
typedef __attribute__((address_space(3))) _Float16 lds_h;
typedef __attribute__((address_space(3))) v8h_t    lds_v8h;

#define ACT_NONE 0
#define ACT_RELU 1

// ---------------------------------------------------------------------------
// WMMA tile GEMM: out[64 x NOUT] (f16, LDS) = act[64 x K] (f16, LDS) * wf^T
// wf is row-major f16 [NOUT x K] in global (workspace), + bias (f32), + act.
// 256 threads = 8 waves: 4 M-tiles x 2 N-tiles of 16x16, loop over N passes.
// ---------------------------------------------------------------------------
template <int K, int NOUT, int ACT>
__device__ __forceinline__ void gemm_tile(const lds_h* act, int lda,
                                          const _Float16* __restrict__ wf,
                                          const float* __restrict__ bias,
                                          lds_h* out, int ldo, int coloff)
{
    const int tid  = threadIdx.x;
    const int wave = tid >> 5;
    const int lane = tid & 31;
    const int hsel = lane >> 4;   // lane half: selects K sub-blocks / M offset
    const int l16  = lane & 15;
    const int m0   = (wave >> 1) * 16;
    const int npass = NOUT >> 5;
    // A fragment base: row m0+l16; halves read K {0..15} or {16..31}
    const lds_h* abase = act + (m0 + l16) * lda + hsel * 16;
#pragma unroll
    for (int p = 0; p < npass; ++p) {
        const int n0 = p * 32 + (wave & 1) * 16;
        // B fragment base: column n0+l16, 16 contiguous k values per lane
        const _Float16* bbase = wf + (n0 + l16) * K + hsel * 16;
        v8f c = {};
#pragma unroll
        for (int k0 = 0; k0 < K; k0 += 32) {
            v8h_t alo = *(const lds_v8h*)(abase + k0);
            v8h_t ahi = *(const lds_v8h*)(abase + k0 + 8);
            v16h a;
#pragma unroll
            for (int i = 0; i < 8; ++i) { a[i] = alo[i]; a[8 + i] = ahi[i]; }
            v16h b = *(const v16h*)(bbase + k0);
            c = __builtin_amdgcn_wmma_f32_16x16x32_f16(false, a, false, b,
                                                       (short)0, c, false, false);
        }
        const int   ncol = coloff + n0 + l16;
        const float bv   = bias ? bias[n0 + l16] : 0.0f;
        lds_h* obase = out + (m0 + hsel * 8) * ldo + ncol;
#pragma unroll
        for (int v = 0; v < 8; ++v) {
            float val = c[v] + bv;
            if (ACT == ACT_RELU) val = fmaxf(val, 0.0f);
            obase[v * ldo] = (_Float16)val;
        }
    }
}

// ---------------------------------------------------------------------------
// Fused per-point chain: feat_in -> fc1 -> LN -> GELU -> fc2 -> concat(pos)
// -> enc0 -> enc1 -> enc2 (ReLU after each). Result: s1[64 x 128] (f16).
// ---------------------------------------------------------------------------
__device__ __forceinline__ void compute_local(int blk,
                              const float* __restrict__ posA, const float* __restrict__ posB,
                              const float* __restrict__ fA,   const float* __restrict__ fB,
                              const float* __restrict__ b1,   const float* __restrict__ lng,
                              const float* __restrict__ lnb,  const float* __restrict__ b2,
                              const float* __restrict__ eb0,  const float* __restrict__ eb1,
                              const float* __restrict__ eb2,
                              const _Float16* __restrict__ wf1, const _Float16* __restrict__ wf2,
                              const _Float16* __restrict__ we0, const _Float16* __restrict__ we1,
                              const _Float16* __restrict__ we2,
                              lds_h* s0, lds_h* s1)
{
    const int tid   = threadIdx.x;
    const int gbase = blk * 64;

    // 1) feat_in = [|fA-fB| , fA*fB]  -> s0[64 x 256]
    for (int idx = tid; idx < 64 * 128; idx += 256) {
        const int row = idx >> 7, c = idx & 127;
        const float a = fA[(size_t)(gbase + row) * CFEAT + c];
        const float b = fB[(size_t)(gbase + row) * CFEAT + c];
        s0[row * LDA0 + c]       = (_Float16)fabsf(a - b);
        s0[row * LDA0 + 128 + c] = (_Float16)(a * b);
    }
    __syncthreads();

    // 2) h = feat_in @ w1^T + b1   -> s1[64 x 64]
    gemm_tile<256, 64, ACT_NONE>(s0, LDA0, wf1, b1, s1, LDA1, 0);
    __syncthreads();

    // 3) LayerNorm + exact GELU (per point over 64 channels), in place
    if (tid < 64) {
        float x[64], m = 0.f;
#pragma unroll
        for (int c = 0; c < 64; ++c) { x[c] = (float)s1[tid * LDA1 + c]; m += x[c]; }
        m *= (1.0f / 64.0f);
        float v = 0.f;
#pragma unroll
        for (int c = 0; c < 64; ++c) { const float d = x[c] - m; v += d * d; }
        v *= (1.0f / 64.0f);
        const float inv = rsqrtf(v + EPSLN);
#pragma unroll
        for (int c = 0; c < 64; ++c) {
            float y = (x[c] - m) * inv * lng[c] + lnb[c];
            y = 0.5f * y * (1.0f + erff(y * 0.70710678118654752f));
            s1[tid * LDA1 + c] = (_Float16)y;
        }
    }
    __syncthreads();

    // 4) feat_enc = h @ w2^T + b2 -> s0 cols 4..35 ; pos in cols 0..3, pad to 63
    gemm_tile<64, 32, ACT_NONE>(s1, LDA1, wf2, b2, s0, LDA0, 4);
    if (tid < 64) {
        const int row = tid;
        s0[row * LDA0 + 0] = (_Float16)posA[(size_t)(gbase + row) * 2 + 0];
        s0[row * LDA0 + 1] = (_Float16)posA[(size_t)(gbase + row) * 2 + 1];
        s0[row * LDA0 + 2] = (_Float16)posB[(size_t)(gbase + row) * 2 + 0];
        s0[row * LDA0 + 3] = (_Float16)posB[(size_t)(gbase + row) * 2 + 1];
#pragma unroll
        for (int c = 36; c < 64; ++c) s0[row * LDA0 + c] = (_Float16)0.0f;
    }
    __syncthreads();

    // 5..7) encoder convs (BN scale folded into weights)
    gemm_tile<64, 128, ACT_RELU>(s0, LDA0, we0, eb0, s1, LDA1, 0);
    __syncthreads();
    gemm_tile<128, 128, ACT_RELU>(s1, LDA1, we1, eb1, s0, LDA0, 0);
    __syncthreads();
    gemm_tile<128, 128, ACT_RELU>(s0, LDA0, we2, eb2, s1, LDA1, 0);
    __syncthreads();
}

// ---------------------------------------------------------------------------
__global__ void __launch_bounds__(256) fold_w_kernel(const float* __restrict__ w,
                                                     const float* __restrict__ g,
                                                     _Float16* __restrict__ out,
                                                     int nout, int kin, int kpad)
{
    const int idx = blockIdx.x * 256 + threadIdx.x;
    if (idx >= nout * kpad) return;
    const int n = idx / kpad, k = idx - n * kpad;
    float v = 0.0f;
    if (k < kin) {
        v = w[n * kin + k];
        if (g) v *= g[n] * 0.99999500003749968f;   // 1/sqrt(1+1e-5)
    }
    out[idx] = (_Float16)v;
}

__global__ void __launch_bounds__(256) zero_f32_kernel(float* p, int n)
{
    const int i = blockIdx.x * 256 + threadIdx.x;
    if (i < n) p[i] = 0.0f;
}

// ---------------------------------------------------------------------------
__global__ void __launch_bounds__(256) encode_max_kernel(
    const float* posA, const float* posB, const float* fA, const float* fB,
    const float* b1, const float* lng, const float* lnb, const float* b2,
    const float* eb0, const float* eb1, const float* eb2,
    const _Float16* wf1, const _Float16* wf2, const _Float16* we0,
    const _Float16* we1, const _Float16* we2, float* glob)
{
    __shared__ alignas(32) _Float16 s0raw[64 * LDA0];
    __shared__ alignas(32) _Float16 s1raw[64 * LDA1];
    lds_h* s0 = (lds_h*)s0raw;
    lds_h* s1 = (lds_h*)s1raw;
    compute_local(blockIdx.x, posA, posB, fA, fB, b1, lng, lnb, b2,
                  eb0, eb1, eb2, wf1, wf2, we0, we1, we2, s0, s1);
    const int tid = threadIdx.x;
    const int batch = (blockIdx.x * 64) >> 12;   // 4096 pts per batch
    if (tid < 128) {
        float mx = 0.0f;
        for (int r = 0; r < 64; ++r) mx = fmaxf(mx, (float)s1[r * LDA1 + tid]);
        atomicMax((unsigned int*)&glob[batch * 128 + tid], __float_as_uint(mx));
    }
}

// ---------------------------------------------------------------------------
__global__ void __launch_bounds__(256) head_kernel(
    const float* posA, const float* posB, const float* fA, const float* fB,
    const float* b1, const float* lng, const float* lnb, const float* b2,
    const float* eb0, const float* eb1, const float* eb2,
    const _Float16* wf1, const _Float16* wf2, const _Float16* we0,
    const _Float16* we1, const _Float16* we2,
    const float* glob, const _Float16* wh0, const _Float16* wh1,
    const float* hb0, const float* hb1,
    const float* hw2, const float* hb2, float* wts)
{
    __shared__ alignas(32) _Float16 s0raw[64 * LDA0];
    __shared__ alignas(32) _Float16 s1raw[64 * LDA1];
    lds_h* s0 = (lds_h*)s0raw;
    lds_h* s1 = (lds_h*)s1raw;
    compute_local(blockIdx.x, posA, posB, fA, fB, b1, lng, lnb, b2,
                  eb0, eb1, eb2, wf1, wf2, we0, we1, we2, s0, s1);
    const int tid = threadIdx.x;
    const int gbase = blockIdx.x * 64;
    const int batch = gbase >> 12;
    // fused = [local | glob]
    for (int idx = tid; idx < 64 * 128; idx += 256) {
        const int row = idx >> 7, c = idx & 127;
        s0[row * LDA0 + c]       = (_Float16)s1[row * LDA1 + c];
        s0[row * LDA0 + 128 + c] = (_Float16)glob[batch * 128 + c];
    }
    __syncthreads();
    gemm_tile<256, 128, ACT_RELU>(s0, LDA0, wh0, hb0, s1, LDA1, 0);
    __syncthreads();
    gemm_tile<128, 64, ACT_RELU>(s1, LDA1, wh1, hb1, s0, LDA0, 0);
    __syncthreads();
    if (tid < 64) {
        float acc = hb2[0];
#pragma unroll
        for (int c = 0; c < 64; ++c) acc += (float)s0[tid * LDA0 + c] * hw2[c];
        wts[gbase + tid] = 1.0f / (1.0f + expf(-acc));
    }
}

// ---------------------------------------------------------------------------
// Deterministic block-wide sum (wave shuffles + LDS), result to all threads.
// ---------------------------------------------------------------------------
__device__ __forceinline__ float block_sum(float v, float* red8)
{
    for (int off = 16; off; off >>= 1) v += __shfl_down(v, off, 32);
    const int w = threadIdx.x >> 5, l = threadIdx.x & 31;
    __syncthreads();
    if (l == 0) red8[w] = v;
    __syncthreads();
    float s = 0.f;
#pragma unroll
    for (int i = 0; i < 8; ++i) s += red8[i];
    return s;
}

__device__ __forceinline__ void mm3(const float A[3][3], const float B[3][3], float C[3][3])
{
    for (int i = 0; i < 3; ++i)
        for (int j = 0; j < 3; ++j)
            C[i][j] = A[i][0] * B[0][j] + A[i][1] * B[1][j] + A[i][2] * B[2][j];
}

__global__ void __launch_bounds__(256) dlt_kernel(const float* __restrict__ posA,
                                                  const float* __restrict__ posB,
                                                  const float* __restrict__ wts,
                                                  float* __restrict__ out)
{
    const int b = blockIdx.x, tid = threadIdx.x;
    __shared__ float red[8];
    __shared__ float accS[44];
    const float* pA = posA + (size_t)b * NPTS * 2;
    const float* pB = posB + (size_t)b * NPTS * 2;
    const float* ww = wts + (size_t)b * NPTS;

    // pass 1: centroids
    float sAx = 0, sAy = 0, sBx = 0, sBy = 0;
    for (int i = tid; i < NPTS; i += 256) {
        sAx += pA[i * 2]; sAy += pA[i * 2 + 1];
        sBx += pB[i * 2]; sBy += pB[i * 2 + 1];
    }
    const float cAx = block_sum(sAx, red) * (1.0f / NPTS);
    const float cAy = block_sum(sAy, red) * (1.0f / NPTS);
    const float cBx = block_sum(sBx, red) * (1.0f / NPTS);
    const float cBy = block_sum(sBy, red) * (1.0f / NPTS);

    // pass 2: mean distances -> Hartley scales
    float dA = 0, dB = 0;
    for (int i = tid; i < NPTS; i += 256) {
        const float ax = pA[i * 2] - cAx, ay = pA[i * 2 + 1] - cAy;
        const float bx = pB[i * 2] - cBx, by = pB[i * 2 + 1] - cBy;
        dA += sqrtf(fmaxf(ax * ax + ay * ay, 0.f));
        dB += sqrtf(fmaxf(bx * bx + by * by, 0.f));
    }
    const float mA = block_sum(dA, red) * (1.0f / NPTS);
    const float mB = block_sum(dB, red) * (1.0f / NPTS);
    const float scA = (mA < 0.001f) ? 1.0f : 1.41421356237f / fmaxf(mA, 0.001f);
    const float scB = (mB < 0.001f) ? 1.0f : 1.41421356237f / fmaxf(mB, 0.001f);

    // pass 3: AtWA (upper triangle, 36) and AtWb (8)
    float accA[36], accB8[8];
#pragma unroll
    for (int j = 0; j < 36; ++j) accA[j] = 0.f;
#pragma unroll
    for (int j = 0; j < 8; ++j) accB8[j] = 0.f;
    for (int i = tid; i < NPTS; i += 256) {
        const float sx = (pA[i * 2] - cAx) * scA, sy = (pA[i * 2 + 1] - cAy) * scA;
        const float dx = (pB[i * 2] - cBx) * scB, dy = (pB[i * 2 + 1] - cBy) * scB;
        const float w = ww[i];
        const float r1[8] = { sx, sy, 1.f, 0.f, 0.f, 0.f, -dx * sx, -dx * sy };
        const float r2[8] = { 0.f, 0.f, 0.f, sx, sy, 1.f, -dy * sx, -dy * sy };
        int idx = 0;
#pragma unroll
        for (int k = 0; k < 8; ++k) {
#pragma unroll
            for (int l = k; l < 8; ++l) { accA[idx] += w * (r1[k] * r1[l] + r2[k] * r2[l]); ++idx; }
            accB8[k] += w * (r1[k] * dx + r2[k] * dy);
        }
    }
#pragma unroll
    for (int j = 0; j < 36; ++j) { const float s = block_sum(accA[j], red); if (tid == 0) accS[j] = s; }
#pragma unroll
    for (int j = 0; j < 8; ++j) { const float s = block_sum(accB8[j], red); if (tid == 0) accS[36 + j] = s; }
    __syncthreads();

    if (tid == 0) {
        float M[8][8], rhs[8], h8[8];
        int idx = 0;
        for (int k = 0; k < 8; ++k)
            for (int l = k; l < 8; ++l) { M[k][l] = accS[idx]; M[l][k] = accS[idx]; ++idx; }
        for (int k = 0; k < 8; ++k) M[k][k] += REGL;
        for (int k = 0; k < 8; ++k) rhs[k] = accS[36 + k];
        rhs[0] += REGL; rhs[4] += REGL;
        // Gaussian elimination with partial pivoting
        for (int c = 0; c < 8; ++c) {
            int piv = c; float mx = fabsf(M[c][c]);
            for (int r = c + 1; r < 8; ++r) { const float a = fabsf(M[r][c]); if (a > mx) { mx = a; piv = r; } }
            if (piv != c) {
                for (int j = 0; j < 8; ++j) { const float t = M[c][j]; M[c][j] = M[piv][j]; M[piv][j] = t; }
                const float t = rhs[c]; rhs[c] = rhs[piv]; rhs[piv] = t;
            }
            const float inv = 1.0f / M[c][c];
            for (int r = c + 1; r < 8; ++r) {
                const float f = M[r][c] * inv;
                for (int j = c; j < 8; ++j) M[r][j] -= f * M[c][j];
                rhs[r] -= f * rhs[c];
            }
        }
        for (int r = 7; r >= 0; --r) {
            float s = rhs[r];
            for (int j = r + 1; j < 8; ++j) s -= M[r][j] * h8[j];
            h8[r] = s / M[r][r];
        }
        bool fin = true;
        for (int k = 0; k < 8; ++k) fin = fin && isfinite(h8[k]);
        if (!fin) { h8[0] = 1; h8[1] = 0; h8[2] = 0; h8[3] = 0; h8[4] = 1; h8[5] = 0; h8[6] = 0; h8[7] = 0; }

        const float Hn[3][3]  = { { h8[0], h8[1], h8[2] }, { h8[3], h8[4], h8[5] }, { h8[6], h8[7], 1.f } };
        const float Ts[3][3]  = { { scA, 0.f, -scA * cAx }, { 0.f, scA, -scA * cAy }, { 0.f, 0.f, 1.f } };
        const float s_dst = fmaxf(scB, 1e-6f);
        const float cxd = (scB * cBx) / s_dst;
        const float cyd = (scB * cBy) / s_dst;
        const float is = 1.0f / s_dst;
        const float Tdi[3][3] = { { is, 0.f, cxd }, { 0.f, is, cyd }, { 0.f, 0.f, 1.f } };
        float Tmp[3][3], H[3][3];
        mm3(Hn, Ts, Tmp);
        mm3(Tdi, Tmp, H);
        const float d1 = fmaxf(fabsf(H[2][2]), 1e-8f);
        for (int i = 0; i < 3; ++i) for (int j = 0; j < 3; ++j) H[i][j] /= d1;
        const float h33 = H[2][2];
        const float sgn = (h33 > 0.f) ? 1.f : ((h33 < 0.f) ? -1.f : 1.f);
        const float d2 = fmaxf(fabsf(h33), 1e-8f) * sgn;
        for (int i = 0; i < 3; ++i) for (int j = 0; j < 3; ++j) H[i][j] /= d2;
        bool hf = true;
        for (int i = 0; i < 3; ++i) for (int j = 0; j < 3; ++j) hf = hf && isfinite(H[i][j]);
        const float a33 = fabsf(H[2][2]);
        const bool valid = hf && (a33 > 1e-4f) && (a33 < 1e4f);
        for (int i = 0; i < 3; ++i)
            for (int j = 0; j < 3; ++j)
                out[b * 9 + i * 3 + j] = valid ? H[i][j] : ((i == j) ? 1.0f : 0.0f);
    }
}

// ---------------------------------------------------------------------------
extern "C" void kernel_launch(void* const* d_in, const int* in_sizes, int n_in,
                              void* d_out, int out_size, void* d_ws, size_t ws_size,
                              hipStream_t stream)
{
    const float* pos_A   = (const float*)d_in[0];
    const float* pos_B   = (const float*)d_in[1];
    const float* feat_A  = (const float*)d_in[2];
    const float* feat_B  = (const float*)d_in[3];
    const float* fc_w1   = (const float*)d_in[4];
    const float* fc_b1   = (const float*)d_in[5];
    const float* fc_ln_g = (const float*)d_in[6];
    const float* fc_ln_b = (const float*)d_in[7];
    const float* fc_w2   = (const float*)d_in[8];
    const float* fc_b2   = (const float*)d_in[9];
    const float* enc_w0  = (const float*)d_in[10];
    const float* enc_g0  = (const float*)d_in[11];
    const float* enc_b0  = (const float*)d_in[12];
    const float* enc_w1  = (const float*)d_in[13];
    const float* enc_g1  = (const float*)d_in[14];
    const float* enc_b1  = (const float*)d_in[15];
    const float* enc_w2  = (const float*)d_in[16];
    const float* enc_g2  = (const float*)d_in[17];
    const float* enc_b2  = (const float*)d_in[18];
    const float* head_w0 = (const float*)d_in[19];
    const float* head_g0 = (const float*)d_in[20];
    const float* head_b0 = (const float*)d_in[21];
    const float* head_w1 = (const float*)d_in[22];
    const float* head_g1 = (const float*)d_in[23];
    const float* head_b1 = (const float*)d_in[24];
    const float* head_w2 = (const float*)d_in[25];
    const float* head_b2 = (const float*)d_in[26];

    char* ws = (char*)d_ws;
    _Float16* wf1 = (_Float16*)(ws + 0);        // 64x256
    _Float16* wf2 = (_Float16*)(ws + 32768);    // 32x64
    _Float16* we0 = (_Float16*)(ws + 36864);    // 128x64 (pad 36->64)
    _Float16* we1 = (_Float16*)(ws + 53248);    // 128x128
    _Float16* we2 = (_Float16*)(ws + 86016);    // 128x128
    _Float16* wh0 = (_Float16*)(ws + 118784);   // 128x256
    _Float16* wh1 = (_Float16*)(ws + 184320);   // 64x128
    float*    glob = (float*)(ws + 200704);     // 64x128
    float*    wts  = (float*)(ws + 233472);     // 64x4096

    fold_w_kernel<<<64, 256, 0, stream>>>(fc_w1, nullptr, wf1, 64, 256, 256);
    fold_w_kernel<<<8, 256, 0, stream>>>(fc_w2, nullptr, wf2, 32, 64, 64);
    fold_w_kernel<<<32, 256, 0, stream>>>(enc_w0, enc_g0, we0, 128, 36, 64);
    fold_w_kernel<<<64, 256, 0, stream>>>(enc_w1, enc_g1, we1, 128, 128, 128);
    fold_w_kernel<<<64, 256, 0, stream>>>(enc_w2, enc_g2, we2, 128, 128, 128);
    fold_w_kernel<<<128, 256, 0, stream>>>(head_w0, head_g0, wh0, 128, 256, 256);
    fold_w_kernel<<<32, 256, 0, stream>>>(head_w1, head_g1, wh1, 64, 128, 128);
    zero_f32_kernel<<<32, 256, 0, stream>>>(glob, 64 * 128);

    encode_max_kernel<<<4096, 256, 0, stream>>>(
        pos_A, pos_B, feat_A, feat_B, fc_b1, fc_ln_g, fc_ln_b, fc_b2,
        enc_b0, enc_b1, enc_b2, wf1, wf2, we0, we1, we2, glob);

    head_kernel<<<4096, 256, 0, stream>>>(
        pos_A, pos_B, feat_A, feat_B, fc_b1, fc_ln_g, fc_ln_b, fc_b2,
        enc_b0, enc_b1, enc_b2, wf1, wf2, we0, we1, we2,
        glob, wh0, wh1, head_b0, head_b1, head_w2, head_b2, wts);

    dlt_kernel<<<64, 256, 0, stream>>>(pos_A, pos_B, wts, (float*)d_out);
}
